// BahdanauAttention_34076270526621
// MI455X (gfx1250) — matrix operand order repaired and checked by reading, not compiled, not run on hardware
//
#include <hip/hip_runtime.h>
#include <hip/hip_bf16.h>

// ---------------------------------------------------------------------------
// Bahdanau attention for MI455X (gfx1250, wave32, WMMA).
//   values:[B,S,D] query:[B,D] W1:[D,U] b1:[U] W2:[D,U] b2:[U] V:[U,1] bV:[1]
//   out = concat(context[B,D], attn[B,S])
// Big GEMM ([B*S,D]x[D,U]) on v_wmma_f32_16x16x32_bf16 with:
//   - W1T staged once per workgroup in LDS (bank-conflict-free padded rows)
//   - 32 rows per wave (each LDS B-fragment feeds two WMMAs from registers)
//   - double-buffered LDS B fragments (loads for step t+1 in flight under
//     the WMMAs of step t -> partial s_wait_dscnt instead of full drains)
//   - gfx1250 hardware v_tanh_f32 epilogue
// ---------------------------------------------------------------------------

typedef __attribute__((ext_vector_type(16))) __bf16          v16bf;
typedef __attribute__((ext_vector_type(8)))  float           v8f;
typedef __attribute__((ext_vector_type(8)))  unsigned short  ushort8;

#define B_  16
#define S_  4096
#define D_  256
#define U_  256
#define ROWS_PER_WG 256            // 8 waves * 32 rows each
#define S_TILES (S_ / ROWS_PER_WG) // 16
#define LDS_PITCH 264              // 256 + 8 pad ushorts -> 528B row stride (4 banks)
#define LDS_BYTES (U_ * LDS_PITCH * sizeof(unsigned short))   // 135168 B (<320KB WGP LDS)

static __device__ __forceinline__ unsigned short f32_to_bf16_bits(float f) {
  union { float f; unsigned u; } v; v.f = f;
  unsigned u = v.u;
  unsigned r = u + 0x7FFFu + ((u >> 16) & 1u);   // round-to-nearest-even
  return (unsigned short)(r >> 16);
}

// gfx1250 hardware tanh transcendental (v_tanh_f32) when available.
static __device__ __forceinline__ float fast_tanh(float x) {
#if __has_builtin(__builtin_amdgcn_tanhf)
  return __builtin_amdgcn_tanhf(x);
#elif __has_builtin(__builtin_amdgcn_tanh_f32)
  return __builtin_amdgcn_tanh_f32(x);
#else
  return tanhf(x);
#endif
}

// ---- K0: W1 [D,U] fp32 -> W1T [U,D] bf16 (transposed for contiguous B frags)
__global__ void k_convert_w1t(const float* __restrict__ W1,
                              unsigned short* __restrict__ W1T) {
  int idx = blockIdx.x * blockDim.x + threadIdx.x;   // 0 .. D*U-1
  int d = idx & (D_ - 1);
  int u = idx >> 8;
  W1T[u * D_ + d] = f32_to_bf16_bits(W1[d * U_ + u]);
}

// ---- K1: qpv = query@W2 + b2 + b1 (b1 folded in); zero context output
__global__ void k_qproj(const float* __restrict__ query,
                        const float* __restrict__ W2,
                        const float* __restrict__ b1,
                        const float* __restrict__ b2,
                        float* __restrict__ qpv,
                        float* __restrict__ ctx_out) {
  int b = blockIdx.x;
  int u = threadIdx.x;
  float acc = b1[u] + b2[u];
  const float* q = query + b * D_;
  #pragma unroll 4
  for (int d = 0; d < D_; ++d)
    acc = fmaf(q[d], W2[d * U_ + u], acc);
  qpv[b * U_ + u] = acc;
  ctx_out[b * D_ + u] = 0.0f;   // k_context accumulates atomically
}

// A fragment loader: 16x32 bf16 per k-step, per ISA layout:
//   lanes 0-15 : M=lane,    K = {0..7} U {16..23} (of each 32-block)
//   lanes 16-31: M=lane-16, K = {8..15} U {24..31}
static __device__ __forceinline__ void load_a_frags(const float* __restrict__ vrow,
                                                    int khalf, v16bf* frag) {
  #pragma unroll
  for (int kt = 0; kt < 8; ++kt) {
    const int k0 = kt * 32;
    union { v16bf v; unsigned short s[16]; } ua;
    const float4* p0 = (const float4*)(vrow + k0 + khalf);
    const float4* p1 = (const float4*)(vrow + k0 + 16 + khalf);
    float4 f0 = p0[0], f1 = p0[1], f2 = p1[0], f3 = p1[1];
    ua.s[0]  = f32_to_bf16_bits(f0.x); ua.s[1]  = f32_to_bf16_bits(f0.y);
    ua.s[2]  = f32_to_bf16_bits(f0.z); ua.s[3]  = f32_to_bf16_bits(f0.w);
    ua.s[4]  = f32_to_bf16_bits(f1.x); ua.s[5]  = f32_to_bf16_bits(f1.y);
    ua.s[6]  = f32_to_bf16_bits(f1.z); ua.s[7]  = f32_to_bf16_bits(f1.w);
    ua.s[8]  = f32_to_bf16_bits(f2.x); ua.s[9]  = f32_to_bf16_bits(f2.y);
    ua.s[10] = f32_to_bf16_bits(f2.z); ua.s[11] = f32_to_bf16_bits(f2.w);
    ua.s[12] = f32_to_bf16_bits(f3.x); ua.s[13] = f32_to_bf16_bits(f3.y);
    ua.s[14] = f32_to_bf16_bits(f3.z); ua.s[15] = f32_to_bf16_bits(f3.w);
    frag[kt] = ua.v;
  }
}

// ---- K2: WMMA GEMM + tanh + dot(V) fused -> raw scores into attn region
__global__ void __launch_bounds__(256)
k_scores(const float* __restrict__ values,
         const unsigned short* __restrict__ W1T,
         const float* __restrict__ qpv,
         const float* __restrict__ V,
         const float* __restrict__ bV,
         float* __restrict__ scores /* d_out + B_*D_, laid out [B,S] */) {
  extern __shared__ unsigned short ldsW[];     // [U_][LDS_PITCH]

  const int wg   = blockIdx.x;                 // B_ * S_TILES blocks
  const int b    = wg / S_TILES;
  const int tile = wg % S_TILES;
  const int wave = threadIdx.x >> 5;
  const int lane = threadIdx.x & 31;

  // ---- Stage W1T (bf16, row u contiguous) into padded LDS: one u-row/thread.
  {
    const uint4* src = (const uint4*)(W1T + (size_t)threadIdx.x * D_);
    uint4* dst = (uint4*)(ldsW + (size_t)threadIdx.x * LDS_PITCH);
    #pragma unroll
    for (int i = 0; i < D_ / 8; ++i) dst[i] = src[i];
  }
  __syncthreads();

  // ---- Two 16-row strips per wave.
  const int m0    = tile * ROWS_PER_WG + wave * 32;
  const int khalf = (lane >> 4) * 8;
  const int mA    = m0 + (lane & 15);
  const int mB    = mA + 16;

  v16bf aFragA[8], aFragB[8];
  load_a_frags(values + ((size_t)b * S_ + mA) * D_, khalf, aFragA);
  load_a_frags(values + ((size_t)b * S_ + mB) * D_, khalf, aFragB);

  // B fragment source in LDS: column u = n0+(lane&15), K-half by lane[4].
  const unsigned short* lwbase = ldsW + (size_t)(lane & 15) * LDS_PITCH + (lane >> 4) * 8;

  float sAccA[8], sAccB[8];
  #pragma unroll
  for (int r = 0; r < 8; ++r) { sAccA[r] = 0.0f; sAccB[r] = 0.0f; }

  const float* qpb = qpv + b * U_;

  // Double-buffered B fragments: loads for step t+1 in flight under WMMAs
  // of step t. 8 k-steps per nt keeps the parity aligned across nt.
  union BFrag { v16bf v; ushort8 h[2]; };
  BFrag frag[2];
  {
    const ushort8* bp = (const ushort8*)lwbase;       // nt=0, kt=0
    frag[0].h[0] = bp[0];
    frag[0].h[1] = bp[2];
  }

  #pragma unroll 1
  for (int nt = 0; nt < 16; ++nt) {
    const unsigned short* lw  = lwbase + (size_t)(nt * 16) * LDS_PITCH;
    const unsigned short* lwn = lwbase + (size_t)(((nt + 1) & 15) * 16) * LDS_PITCH;
    v8f accA = {}, accB = {};
    #pragma unroll
    for (int kt = 0; kt < 8; ++kt) {
      const v16bf bcur = frag[kt & 1].v;
      // Prefetch next fragment (next kt, or kt=0 of next nt).
      const ushort8* bp = (kt < 7) ? (const ushort8*)(lw + (kt + 1) * 32)
                                   : (const ushort8*)lwn;
      frag[(kt + 1) & 1].h[0] = bp[0];
      frag[(kt + 1) & 1].h[1] = bp[2];
      accA = __builtin_amdgcn_wmma_f32_16x16x32_bf16(
          false, aFragA[kt], false, bcur, (short)0, accA, false, false);
      accB = __builtin_amdgcn_wmma_f32_16x16x32_bf16(
          false, aFragB[kt], false, bcur, (short)0, accB, false, false);
    }
    // C/D layout: VGPR r, lanes 0-15 -> (M=r, N=lane); lanes 16-31 -> (M=r+8)
    const int n0 = nt * 16;
    const float qpn = qpb[n0 + (lane & 15)];
    const float vn  = V[n0 + (lane & 15)];
    #pragma unroll
    for (int r = 0; r < 8; ++r) {
      sAccA[r] = fmaf(fast_tanh(accA[r] + qpn), vn, sAccA[r]);
      sAccB[r] = fmaf(fast_tanh(accB[r] + qpn), vn, sAccB[r]);
    }
  }

  // Reduce over N (16-lane halves); each half keeps its own row set.
  #pragma unroll
  for (int r = 0; r < 8; ++r) {
    float sa = sAccA[r], sb = sAccB[r];
    sa += __shfl_xor(sa, 1, 32);  sb += __shfl_xor(sb, 1, 32);
    sa += __shfl_xor(sa, 2, 32);  sb += __shfl_xor(sb, 2, 32);
    sa += __shfl_xor(sa, 4, 32);  sb += __shfl_xor(sb, 4, 32);
    sa += __shfl_xor(sa, 8, 32);  sb += __shfl_xor(sb, 8, 32);
    sAccA[r] = sa; sAccB[r] = sb;
  }
  if ((lane & 15) == 0) {
    const float bv = bV[0];
    const int rbase = m0 + ((lane >> 4) << 3);   // rows m0..m0+7 / m0+8..m0+15
    #pragma unroll
    for (int r = 0; r < 8; ++r) {
      scores[(size_t)b * S_ + rbase + r]      = sAccA[r] + bv;
      scores[(size_t)b * S_ + rbase + 16 + r] = sAccB[r] + bv;
    }
  }
}

// ---- K3: in-place softmax over S per batch
__global__ void k_softmax(float* __restrict__ attn) {
  const int b = blockIdx.x, tid = threadIdx.x;
  __shared__ float red[256];
  float* s = attn + (size_t)b * S_;

  float loc[16];
  float mx = -3.0e38f;
  #pragma unroll
  for (int i = 0; i < 16; ++i) { loc[i] = s[tid + i * 256]; mx = fmaxf(mx, loc[i]); }
  red[tid] = mx; __syncthreads();
  for (int off = 128; off > 0; off >>= 1) {
    if (tid < off) red[tid] = fmaxf(red[tid], red[tid + off]);
    __syncthreads();
  }
  mx = red[0]; __syncthreads();

  float sum = 0.0f;
  #pragma unroll
  for (int i = 0; i < 16; ++i) { loc[i] = __expf(loc[i] - mx); sum += loc[i]; }
  red[tid] = sum; __syncthreads();
  for (int off = 128; off > 0; off >>= 1) {
    if (tid < off) red[tid] += red[tid + off];
    __syncthreads();
  }
  const float inv = 1.0f / red[0];
  #pragma unroll
  for (int i = 0; i < 16; ++i) s[tid + i * 256] = loc[i] * inv;
}

// ---- K4: context[b,d] = sum_s attn[b,s] * values[b,s,d]
__global__ void k_context(const float* __restrict__ values,
                          const float* __restrict__ attn,
                          float* __restrict__ ctx) {
  const int b = blockIdx.x;
  const int s0 = blockIdx.y * 512;
  const int tid = threadIdx.x;   // d
  __shared__ float wsh[512];
  wsh[tid]       = attn[(size_t)b * S_ + s0 + tid];
  wsh[tid + 256] = attn[(size_t)b * S_ + s0 + tid + 256];
  __syncthreads();
  float acc = 0.0f;
  const float* vp = values + ((size_t)b * S_ + s0) * D_ + tid;
  #pragma unroll 8
  for (int s = 0; s < 512; ++s)
    acc = fmaf(wsh[s], vp[(size_t)s * D_], acc);
  atomicAdd(&ctx[b * D_ + tid], acc);
}

// ---------------------------------------------------------------------------
extern "C" void kernel_launch(void* const* d_in, const int* in_sizes, int n_in,
                              void* d_out, int out_size, void* d_ws, size_t ws_size,
                              hipStream_t stream) {
  const float* values = (const float*)d_in[0];
  const float* query  = (const float*)d_in[1];
  const float* W1     = (const float*)d_in[2];
  const float* b1     = (const float*)d_in[3];
  const float* W2     = (const float*)d_in[4];
  const float* b2     = (const float*)d_in[5];
  const float* V      = (const float*)d_in[6];
  const float* bV     = (const float*)d_in[7];

  float* out  = (float*)d_out;
  float* ctx  = out;              // [B, D]
  float* attn = out + B_ * D_;    // [B, S]

  unsigned short* W1T = (unsigned short*)d_ws;                       // 128 KB
  float* qpv = (float*)((char*)d_ws + (size_t)D_ * U_ * sizeof(unsigned short));

  k_convert_w1t<<<(D_ * U_) / 256, 256, 0, stream>>>(W1, W1T);
  k_qproj<<<B_, 256, 0, stream>>>(query, W2, b1, b2, qpv, ctx);
  k_scores<<<B_ * S_TILES, 256, (unsigned)LDS_BYTES, stream>>>(values, W1T, qpv, V, bV, attn);
  k_softmax<<<B_, 256, 0, stream>>>(attn);
  dim3 g(B_, S_ / 512);
  k_context<<<g, 256, 0, stream>>>(values, attn, ctx);
}
